// PolicyGeneratedResidualConvolutionLayer_22892175687863
// MI455X (gfx1250) — matrix operand order repaired and checked by reading, not compiled, not run on hardware
//
#include <hip/hip_runtime.h>

// MI455X / gfx1250: per-sample conv3x3 + residual via bf16x3-split WMMA implicit GEMM.
// out[b] = x[b] + W[b](128x1152) @ im2col(x[b])(1152x4096)
// v5: zero-halo padded input pack -> unconditional, uniform async B staging
//     (sound s_wait_asynccnt<=4 with triple-buffered LDS); A (weights) double-
//     buffered in registers from pre-packed global; N=128 tile (2 image rows).

typedef __attribute__((ext_vector_type(16))) __bf16 v16bf;
typedef __attribute__((ext_vector_type(8)))  float  v8f;

#define NB 16
#define NC 128
#define NH 64
#define NW 64

#define APK_BYTES (16ull * 9 * 4 * 2 * 128 * 32 * 2)        // 9,437,184
// xpk: [b][hh2 0..65][ww2 0..65][hl 0..1][ci 0..127] bf16 halves
#define XPK_HALVES (16ull * 66 * 66 * 2 * 128)

__device__ __forceinline__ unsigned short bf16hi(float f) {
    unsigned u = __float_as_uint(f);
    unsigned r = u + 0x7fffu + ((u >> 16) & 1u);
    return (unsigned short)(r >> 16);
}

__device__ __forceinline__ unsigned lds_addr(const void* p) {
    // flat LDS-aperture address: low 32 bits == LDS byte offset (ISA 10.2)
    return (unsigned)(unsigned long long)(uintptr_t)p;
}

__device__ __forceinline__ void async_b128(unsigned lds_off, const void* g) {
    asm volatile("global_load_async_to_lds_b128 %0, %1, off"
                 :: "v"(lds_off), "v"((unsigned long long)(uintptr_t)g)
                 : "memory");
}

__device__ __forceinline__ void wait_async_le4() {
    asm volatile("s_wait_asynccnt 0x4" ::: "memory");
}

__device__ __forceinline__ void wait_async0() {
    asm volatile("s_wait_asynccnt 0x0" ::: "memory");
}

// ---------------------------------------------------------------------------
// Prepass 1: weight[b][co][ci][kh][kw] (f32) ->
//   apk[b][tap][kc][hl][co][32 halves]  (bf16 hi / lo, plain row layout)
// ---------------------------------------------------------------------------
__global__ __launch_bounds__(256) void prep_weights(const float* __restrict__ w,
                                                    unsigned short* __restrict__ apk) {
    __shared__ float tile[32 * 288];  // 32 co rows x (32 ci * 9 taps) f32
    int bs  = blockIdx.x;             // 16 b * 4 kc * 4 co-quarters = 256 blocks
    int b   = bs >> 4;
    int kc  = (bs >> 2) & 3;
    int coq = bs & 3;
    int co0 = coq * 32;
    int ci0 = kc * 32;
    int t   = threadIdx.x;

    const float* src = w + (size_t)(b * 128 + co0) * 1152 + ci0 * 9;
    for (int idx = t; idx < 32 * 288; idx += 256) {
        int r = idx / 288, c = idx - r * 288;
        tile[r * 288 + c] = src[(size_t)r * 1152 + c];
    }
    __syncthreads();

    for (int rr = t; rr < 576; rr += 256) {
        int hl   = rr & 1;
        int co_r = (rr >> 1) & 31;
        int tap  = rr >> 6;
        int co   = co0 + co_r;
        unsigned wq[8];
#pragma unroll
        for (int kk = 0; kk < 32; ++kk) {
            float f = tile[co_r * 288 + kk * 9 + tap];
            unsigned short hi = bf16hi(f);
            unsigned short v;
            if (hl == 0) {
                v = hi;
            } else {
                float fh = __uint_as_float(((unsigned)hi) << 16);
                v = bf16hi(f - fh);
            }
            if (kk & 1) wq[kk >> 1] |= ((unsigned)v) << 16;
            else        wq[kk >> 1]  = (unsigned)v;
        }
        size_t rowHalves = ((((size_t)(b * 9 + tap) * 4 + kc) * 2 + hl) * 128 + co) * 32;
        uint4* dst = (uint4*)(apk + rowHalves);
        dst[0] = make_uint4(wq[0], wq[1], wq[2], wq[3]);
        dst[1] = make_uint4(wq[4], wq[5], wq[6], wq[7]);
    }
}

// ---------------------------------------------------------------------------
// Prepass 2a: zero the 1-pixel halo of xpk (4160 positions x 512 B).
// ---------------------------------------------------------------------------
__global__ __launch_bounds__(256) void zero_halo(unsigned short* __restrict__ xpk) {
    int gid = blockIdx.x * 256 + threadIdx.x;  // 520 blocks: 4160 pos * 32 thr
    int pos = gid >> 5;
    int q   = gid & 31;                        // 32 x uint4 = 512 B per position
    if (pos >= 4160) return;
    int b = pos / 260;
    int r = pos - b * 260;
    int hh2, ww2;
    if (r < 66)       { hh2 = 0;       ww2 = r; }
    else if (r < 132) { hh2 = 65;      ww2 = r - 66; }
    else if (r < 196) { hh2 = r - 131; ww2 = 0; }    // 1..64
    else              { hh2 = r - 195; ww2 = 65; }   // 1..64
    size_t baseHalves = ((size_t)((b * 66 + hh2) * 66 + ww2)) * 256;
    uint4* dst = (uint4*)(xpk + baseHalves);
    dst[q] = make_uint4(0, 0, 0, 0);
}

// ---------------------------------------------------------------------------
// Prepass 2b: input[b][ci][h][w] (f32) -> xpk[b][h+1][w+1][hl][ci] (bf16 hi/lo)
// Channels-last + zero halo so main-kernel B staging is a branch-free copy.
// ---------------------------------------------------------------------------
__global__ __launch_bounds__(256) void prep_input(const float* __restrict__ in,
                                                  unsigned short* __restrict__ xpk) {
    __shared__ float tile[128 * 65];  // [ci][w], padded stride (conflict-free transpose)
    int h = blockIdx.x;
    int b = blockIdx.y;
    int t = threadIdx.x;

    const float* src = in + (size_t)b * NC * NH * NW + (size_t)h * NW;
#pragma unroll
    for (int i = 0; i < 32; ++i) {
        int idx = t + i * 256;
        int ci = idx >> 6, w = idx & 63;
        tile[ci * 65 + w] = src[(size_t)ci * (NH * NW) + w];
    }
    __syncthreads();

    int rowid = t >> 1;      // w*2 + hl
    int half  = t & 1;       // which 64 ci
    int w  = rowid >> 1;
    int hl = rowid & 1;
    unsigned wq[32];
#pragma unroll
    for (int k = 0; k < 64; ++k) {
        int ci = half * 64 + k;
        float f = tile[ci * 65 + w];
        unsigned short hi = bf16hi(f);
        unsigned short v;
        if (hl == 0) {
            v = hi;
        } else {
            float fh = __uint_as_float(((unsigned)hi) << 16);
            v = bf16hi(f - fh);
        }
        if (k & 1) wq[k >> 1] |= ((unsigned)v) << 16;
        else       wq[k >> 1]  = (unsigned)v;
    }
    size_t rowHalves =
        ((size_t)(((b * 66 + (h + 1)) * 66 + (w + 1)) * 2 + hl)) * 128 + half * 64;
    uint4* dst = (uint4*)(xpk + rowHalves);
#pragma unroll
    for (int q = 0; q < 8; ++q)
        dst[q] = make_uint4(wq[q * 4 + 0], wq[q * 4 + 1], wq[q * 4 + 2], wq[q * 4 + 3]);
}

// ---------------------------------------------------------------------------
// Main kernel: one block per (2 rows, b). M=128 (8 waves x 16), N=128.
// 36 K-chunks (9 taps x 4 ci-chunks of 32). A in registers (2-deep),
// B in LDS (3-deep, uniform async DMA), one barrier per chunk.
// ---------------------------------------------------------------------------
__global__ __launch_bounds__(256) void conv_main(const float* __restrict__ in,
                                                 const unsigned short* __restrict__ apk,
                                                 const unsigned short* __restrict__ xpk,
                                                 float* __restrict__ out) {
    __shared__ alignas(16) unsigned short sB[3][2 * 128 * 40];  // 3 x 20 KB

    int h0   = blockIdx.x * 2;
    int b    = blockIdx.y;
    int t    = threadIdx.x;
    int lane = t & 31, wave = t >> 5;
    int l15  = lane & 15, lh = lane >> 4;
    int m0   = wave * 16;
    int m    = m0 + l15;

    // B staging: thread t owns column n = t&127, 32B sub-row kH = t>>7
    int nS = t & 127, kH = t >> 7;
    int hrS = nS >> 6, wS = nS & 63;

    union U { uint4 q[2]; v16bf v; };

    v8f acc[8];
#pragma unroll
    for (int j = 0; j < 8; ++j) acc[j] = v8f{0.f, 0.f, 0.f, 0.f, 0.f, 0.f, 0.f, 0.f};

    const size_t inBase = (size_t)b * NC * NH * NW;

    // ---- A: per-lane direct global loads into WMMA registers ----
    auto loadA = [&](int it, U& Ahi, U& Alo) {
        int tap = it >> 2, kc = it & 3;
        const unsigned short* cb = apk + ((size_t)((b * 9 + tap) * 4 + kc) << 13);
        const unsigned short* rh = cb + (size_t)m * 32;
        const unsigned short* rl = cb + (size_t)(128 + m) * 32;
        Ahi.q[0] = *(const uint4*)(rh + lh * 16);
        Ahi.q[1] = *(const uint4*)(rh + lh * 16 + 8);
        Alo.q[0] = *(const uint4*)(rl + lh * 16);
        Alo.q[1] = *(const uint4*)(rl + lh * 16 + 8);
    };

    // ---- B: branch-free async DMA (halo supplies zeros) ----
    auto stageB = [&](int it, int buf) {
        int tap = it >> 2, kc = it & 3;
        int hh2 = h0 + hrS + tap / 3;   // (h+dh)+1 in padded coords, 0..65
        int ww2 = wS + tap % 3;         // (w+dw)+1 in padded coords, 0..65
        const unsigned short* srcHi =
            xpk + ((size_t)((b * 66 + hh2) * 66 + ww2) * 2) * 128 + kc * 32 + kH * 16;
        unsigned ldsHi = lds_addr(&sB[buf][nS * 40 + kH * 16]);
        async_b128(ldsHi,      srcHi);            // hi halves (16)
        async_b128(ldsHi + 16, srcHi + 8);
        async_b128(ldsHi + 2 * 128 * 40,      srcHi + 128);  // lo halves
        async_b128(ldsHi + 2 * 128 * 40 + 16, srcHi + 136);
    };

    auto compute = [&](int buf, const U& Ahi, const U& Alo) {
#pragma unroll
        for (int j = 0; j < 8; ++j) {
            int n = j * 16 + l15;
            const unsigned short* bh = &sB[buf][n * 40 + lh * 16];
            const unsigned short* bl = &sB[buf][128 * 40 + n * 40 + lh * 16];
            U Bhi, Blo;
            Bhi.q[0] = *(const uint4*)(bh);
            Bhi.q[1] = *(const uint4*)(bh + 8);
            Blo.q[0] = *(const uint4*)(bl);
            Blo.q[1] = *(const uint4*)(bl + 8);
            acc[j] = __builtin_amdgcn_wmma_f32_16x16x32_bf16(
                false, Ahi.v, false, Bhi.v, (short)0, acc[j], false, false);
            acc[j] = __builtin_amdgcn_wmma_f32_16x16x32_bf16(
                false, Ahi.v, false, Blo.v, (short)0, acc[j], false, false);
            acc[j] = __builtin_amdgcn_wmma_f32_16x16x32_bf16(
                false, Alo.v, false, Bhi.v, (short)0, acc[j], false, false);
        }
    };

    // ---- prologue: B chunks 0,1 in flight; A chunk 0 in regs ----
    U A0hi, A0lo, A1hi, A1lo;
    stageB(0, 0);
    stageB(1, 1);
    loadA(0, A0hi, A0lo);
    wait_async_le4();   // group(0) landed (in-order); group(1) may still fly
    __syncthreads();

    // ---- main pipeline: 1 barrier per chunk ----
    int cur = 0, b2 = 2;  // cur = it%3, b2 = (it+2)%3
    for (int it = 0; it < 36; ++it) {
        if (it + 2 < 36) stageB(it + 2, b2);
        if (it & 1) {
            if (it + 1 < 36) loadA(it + 1, A0hi, A0lo);
            compute(cur, A1hi, A1lo);
        } else {
            if (it + 1 < 36) loadA(it + 1, A1hi, A1lo);
            compute(cur, A0hi, A0lo);
        }
        if (it < 34) wait_async_le4();  // uniform 4-op groups: (it+1) done, (it+2) flying
        else         wait_async0();
        __syncthreads();
        cur = (cur == 2) ? 0 : cur + 1;
        b2  = (b2 == 2) ? 0 : b2 + 1;
    }

    // ---- epilogue: residual add + store (D layout: vgpr i -> rows m0+i / m0+8+i) ----
    int mb = m0 + lh * 8;
#pragma unroll
    for (int j = 0; j < 8; ++j) {
        int n = j * 16 + l15;                       // 0..127
        int hr = n >> 6, w = n & 63;
#pragma unroll
        for (int i = 0; i < 8; ++i) {
            size_t idx = inBase + ((size_t)(mb + i)) * (NH * NW) + (size_t)(h0 + hr) * NW + w;
            out[idx] = in[idx] + acc[j][i];
        }
    }
}

extern "C" void kernel_launch(void* const* d_in, const int* in_sizes, int n_in,
                              void* d_out, int out_size, void* d_ws, size_t ws_size,
                              hipStream_t stream) {
    const float* inp = (const float*)d_in[0];   // [16,128,64,64] f32
    const float* wgt = (const float*)d_in[1];   // [16,128,128,3,3] f32
    float* outp = (float*)d_out;                // [16,128,64,64] f32

    unsigned short* apk = (unsigned short*)d_ws;                       // 9.44 MB
    unsigned short* xpk = (unsigned short*)((char*)d_ws + APK_BYTES);  // 35.68 MB (halo)

    prep_weights<<<256, 256, 0, stream>>>(wgt, apk);
    zero_halo<<<520, 256, 0, stream>>>(xpk);
    prep_input<<<dim3(NH, NB), 256, 0, stream>>>(inp, xpk);
    conv_main<<<dim3(NH / 2, NB), 256, 0, stream>>>(inp, apk, xpk, outp);
}
